// Mamba2Conv_31336081391714
// MI455X (gfx1250) — compile-verified, matrix operand order checked by reading, not run on hardware
//
#include <hip/hip_runtime.h>
#include <hip/hip_bf16.h>

// ---------------------------------------------------------------------------
// Mamba2Conv block for MI455X (gfx1250, wave32, WMMA).
// B=4, C=256, H=W=64 -> L=4096, M=B*L=16384, D_INNER=512, DT_RANK=16, N=16.
// ---------------------------------------------------------------------------

typedef __attribute__((ext_vector_type(16))) _Float16 v16h;
typedef __attribute__((ext_vector_type(8)))  _Float16 v8h;
typedef __attribute__((ext_vector_type(8)))  float    v8f;

#define Lh 4096
#define Cc 256
#define Din 512
#define Mtot 16384   // B * L
#define Etot 1024    // 2 * D_INNER
#define DBLW 48      // dt_rank + 2*N

// ---- per-lane v16h assembly from two 128-bit loads (ISA 16-bit A/B layout:
//      lane<16: K = {0..7, 16..23}; lane>=16: K = {8..15, 24..31}) ------------
__device__ __forceinline__ v16h load_tile16(const _Float16* p) {
  v8h lo = *(const v8h*)(p);        // K = base .. base+7
  v8h hi = *(const v8h*)(p + 16);   // K = base+16 .. base+23
  v16h r;
#pragma unroll
  for (int i = 0; i < 8; ++i) { r[i] = lo[i]; r[i + 8] = hi[i]; }
  return r;
}

// ---- double-buffered WMMA main loop (compile-time K for full unroll) -------
template <int NT, int K>
__device__ __forceinline__ void gemm_loop(const _Float16* __restrict__ ap,
                                          const _Float16* const (&bp)[NT],
                                          v8f (&acc)[NT]) {
  v16h a_cur = load_tile16(ap);
  v16h b_cur[NT];
#pragma unroll
  for (int t = 0; t < NT; ++t) b_cur[t] = load_tile16(bp[t]);

#pragma unroll
  for (int k0 = 32; k0 < K; k0 += 32) {
    // issue next-step loads BEFORE consuming current tiles -> partial waits
    v16h a_nxt = load_tile16(ap + k0);
    v16h b_nxt[NT];
#pragma unroll
    for (int t = 0; t < NT; ++t) b_nxt[t] = load_tile16(bp[t] + k0);
#pragma unroll
    for (int t = 0; t < NT; ++t)
      acc[t] = __builtin_amdgcn_wmma_f32_16x16x32_f16(
          false, a_cur, false, b_cur[t], (short)0, acc[t], false, false);
    a_cur = a_nxt;
#pragma unroll
    for (int t = 0; t < NT; ++t) b_cur[t] = b_nxt[t];
  }
#pragma unroll
  for (int t = 0; t < NT; ++t)
    acc[t] = __builtin_amdgcn_wmma_f32_16x16x32_f16(
        false, a_cur, false, b_cur[t], (short)0, acc[t], false, false);
}

// ---- generic WMMA GEMM: C[m,n] = sum_k A[m,k] * Bw[n,k] --------------------
// A row-major [M,K] f16, Bw row-major [N,K] f16, Cout row-major [M,N] f32.
// One wave per block; each wave computes 16 rows x (16*NT) cols.
template <int NT, int K>
__global__ void wmma_gemm_store(const _Float16* __restrict__ A,
                                const _Float16* __restrict__ Bw,
                                float* __restrict__ Cout,
                                int N) {
  const int m0   = blockIdx.x * 16;
  const int n0   = blockIdx.y * 16 * NT;
  const int lane = threadIdx.x;
  const int half = lane >> 4;     // 0 or 1
  const int r16  = lane & 15;

  const _Float16* ap = A + (size_t)(m0 + r16) * K + half * 8;
  const _Float16* bp[NT];
#pragma unroll
  for (int t = 0; t < NT; ++t)
    bp[t] = Bw + (size_t)(n0 + t * 16 + r16) * K + half * 8;

  v8f acc[NT];
#pragma unroll
  for (int t = 0; t < NT; ++t)
#pragma unroll
    for (int i = 0; i < 8; ++i) acc[t][i] = 0.0f;

  gemm_loop<NT, K>(ap, bp, acc);

#pragma unroll
  for (int t = 0; t < NT; ++t) {
    const int col = n0 + t * 16 + r16;
#pragma unroll
    for (int v = 0; v < 8; ++v) {
      const int row = m0 + v + 8 * half;   // C/D layout: VGPR v, lane half
      Cout[(size_t)row * N + col] = acc[t][v];
    }
  }
}

// ---- out-proj WMMA GEMM with transpose-store to [B,C,H,W] + residual -------
// y f16 [M,512], Wo f16 [256,512], out[b*C*L + c*L + l] = acc + x[...]
__global__ void wmma_outproj(const _Float16* __restrict__ Y,
                             const _Float16* __restrict__ Wo,
                             const float* __restrict__ Xres,
                             float* __restrict__ Out) {
  constexpr int NT = 4;
  const int m0   = blockIdx.x * 16;
  const int n0   = blockIdx.y * 16 * NT;
  const int lane = threadIdx.x;
  const int half = lane >> 4;
  const int r16  = lane & 15;

  const _Float16* ap = Y + (size_t)(m0 + r16) * Din + half * 8;
  const _Float16* bp[NT];
#pragma unroll
  for (int t = 0; t < NT; ++t)
    bp[t] = Wo + (size_t)(n0 + t * 16 + r16) * Din + half * 8;

  v8f acc[NT];
#pragma unroll
  for (int t = 0; t < NT; ++t)
#pragma unroll
    for (int i = 0; i < 8; ++i) acc[t][i] = 0.0f;

  gemm_loop<NT, Din>(ap, bp, acc);

#pragma unroll
  for (int t = 0; t < NT; ++t) {
    const int c = n0 + t * 16 + r16;             // channel
#pragma unroll
    for (int v = 0; v < 8; ++v) {
      const int row = m0 + v + 8 * half;         // token index m
      const int b = row >> 12;                   // / L
      const int l = row & (Lh - 1);
      const size_t idx = ((size_t)b * Cc + c) * Lh + l;
      Out[idx] = acc[t][v] + Xres[idx];
    }
  }
}

// ---- LayerNorm over channels; token-major f16 output -----------------------
__global__ void ln_kernel(const float* __restrict__ X,
                          const float* __restrict__ lw,
                          const float* __restrict__ lb,
                          _Float16* __restrict__ Xn) {
  const int m = blockIdx.x;       // token
  const int c = threadIdx.x;      // 256 threads
  const int b = m >> 12;
  const int l = m & (Lh - 1);
  const float v = X[((size_t)b * Cc + c) * Lh + l];

  __shared__ float s1[Cc];
  __shared__ float s2[Cc];
  s1[c] = v; s2[c] = v * v;
  __syncthreads();
#pragma unroll
  for (int o = 128; o > 0; o >>= 1) {
    if (c < o) { s1[c] += s1[c + o]; s2[c] += s2[c + o]; }
    __syncthreads();
  }
  const float mu  = s1[0] * (1.0f / Cc);
  const float var = s2[0] * (1.0f / Cc) - mu * mu;
  const float r   = rsqrtf(var + 1e-5f);
  Xn[(size_t)m * Cc + c] = (_Float16)((v - mu) * r * lw[c] + lb[c]);
}

// ---- causal depthwise conv (k=4) + SiLU ------------------------------------
__global__ void conv_silu(const float* __restrict__ XZ,
                          const float* __restrict__ cw,
                          const float* __restrict__ cb,
                          float* __restrict__ Xc32,
                          _Float16* __restrict__ Xc16) {
  const int idx = blockIdx.x * blockDim.x + threadIdx.x;   // m*512 + d
  const int d = idx & (Din - 1);
  const int m = idx >> 9;
  const int b = m >> 12;
  const int l = m & (Lh - 1);
  float acc = cb[d];
#pragma unroll
  for (int k = 0; k < 4; ++k) {
    const int ll = l - 3 + k;
    if (ll >= 0)
      acc += XZ[((size_t)(b << 12) + ll) * Etot + d] * cw[d * 4 + k];
  }
  const float s = acc / (1.0f + __expf(-acc));
  Xc32[idx] = s;
  Xc16[idx] = (_Float16)s;
}

// ---- delta = softplus(dt @ W_dt^T + b_dt), K=16 (VALU) ---------------------
__global__ void delta_kernel(const float* __restrict__ Dbl,
                             const float* __restrict__ Wdt,
                             const float* __restrict__ bdt,
                             float* __restrict__ Delta) {
  const int idx = blockIdx.x * blockDim.x + threadIdx.x;   // m*512 + d
  const int d = idx & (Din - 1);
  const int m = idx >> 9;
  float acc = bdt[d];
  const float* dt = Dbl + (size_t)m * DBLW;
#pragma unroll
  for (int r = 0; r < 16; ++r) acc += dt[r] * Wdt[d * 16 + r];
  Delta[idx] = (acc > 20.0f) ? acc : log1pf(__expf(acc));
}

// ---- 16-lane XOR-tree sum via ds_swizzle (group-of-32 encoding) ------------
__device__ __forceinline__ float red16(float v) {
  float t;
  t = __builtin_bit_cast(float, __builtin_amdgcn_ds_swizzle(__builtin_bit_cast(int, v), 0x201f)); v += t; // xor 8
  t = __builtin_bit_cast(float, __builtin_amdgcn_ds_swizzle(__builtin_bit_cast(int, v), 0x101f)); v += t; // xor 4
  t = __builtin_bit_cast(float, __builtin_amdgcn_ds_swizzle(__builtin_bit_cast(int, v), 0x081f)); v += t; // xor 2
  t = __builtin_bit_cast(float, __builtin_amdgcn_ds_swizzle(__builtin_bit_cast(int, v), 0x041f)); v += t; // xor 1
  return v;
}

// ---- selective scan: one 16-lane group per (b,d), n = lane within group ----
__global__ void scan_kernel(const float* __restrict__ Delta,
                            const float* __restrict__ Xc32,
                            const float* __restrict__ Dbl,
                            const float* __restrict__ XZ,
                            const float* __restrict__ Alog,
                            const float* __restrict__ Dp,
                            _Float16* __restrict__ Yout) {
  const int tid  = threadIdx.x;                    // 256
  const int grp  = tid >> 4;
  const int n    = tid & 15;
  const int pair = blockIdx.x * 16 + grp;          // 0 .. 2047
  const int b    = pair >> 9;
  const int d    = pair & (Din - 1);

  const float A  = -__expf(Alog[d * 16 + n]);
  const float Dd = Dp[d];
  float h = 0.0f;
  const size_t mBase = (size_t)b << 12;

  for (int l = 0; l < Lh; ++l) {
    const size_t m  = mBase + l;
    // stream-ahead prefetch of the sequential walk (global_prefetch_b8)
    if (l + 8 < Lh) {
      __builtin_prefetch(&Delta[(m + 8) * Din + d], 0, 1);
      __builtin_prefetch(&Xc32[(m + 8) * Din + d], 0, 1);
      __builtin_prefetch(&Dbl[(m + 8) * DBLW + n], 0, 1);
    }
    const float dlt = Delta[m * Din + d];
    const float u   = Xc32[m * Din + d];
    const float* db = Dbl + m * DBLW;
    const float Bv  = db[16 + n];
    const float Cv  = db[32 + n];
    h = h * __expf(dlt * A) + dlt * u * Bv;
    float y = red16(h * Cv);
    if (n == 0) {
      y += u * Dd;
      const float z = XZ[m * Etot + Din + d];
      y *= z / (1.0f + __expf(-z));
      Yout[m * Din + d] = (_Float16)y;
    }
  }
}

// ---- f32 -> f16 copy -------------------------------------------------------
__global__ void cvt_f16(const float* __restrict__ src, _Float16* __restrict__ dst, int n) {
  const int i = blockIdx.x * blockDim.x + threadIdx.x;
  if (i < n) dst[i] = (_Float16)src[i];
}

// ---------------------------------------------------------------------------
extern "C" void kernel_launch(void* const* d_in, const int* in_sizes, int n_in,
                              void* d_out, int out_size, void* d_ws, size_t ws_size,
                              hipStream_t stream) {
  const float* x      = (const float*)d_in[0];
  const float* ln_w   = (const float*)d_in[1];
  const float* ln_b   = (const float*)d_in[2];
  const float* W_in   = (const float*)d_in[3];   // [1024, 256]
  const float* conv_w = (const float*)d_in[4];   // [512, 4]
  const float* conv_b = (const float*)d_in[5];
  const float* W_x    = (const float*)d_in[6];   // [48, 512]
  const float* W_dt   = (const float*)d_in[7];   // [512, 16]
  const float* b_dt   = (const float*)d_in[8];
  const float* A_log  = (const float*)d_in[9];   // [512, 16]
  const float* Dp     = (const float*)d_in[10];
  const float* W_out  = (const float*)d_in[11];  // [256, 512]
  float* out = (float*)d_out;

  // workspace layout (all offsets 256B-aligned)
  char* ws = (char*)d_ws;
  _Float16* xn16  = (_Float16*)(ws + 0);                         //  8 MB [M,256]
  float*    xz32  = (float*)   (ws + (size_t)8388608);           // 64 MB [M,1024]
  float*    xc32  = (float*)   (ws + (size_t)75497472);          // 32 MB [M,512]
  _Float16* xc16  = (_Float16*)(ws + (size_t)109051904);         // 16 MB [M,512]
  float*    dbl32 = (float*)   (ws + (size_t)125829120);         //  3 MB [M,48]
  float*    dlt32 = (float*)   (ws + (size_t)128974848);         // 32 MB [M,512]
  _Float16* y16   = (_Float16*)(ws + (size_t)162529280);         // 16 MB [M,512]
  _Float16* Win16 = (_Float16*)(ws + (size_t)179306496);
  _Float16* Wx16  = (_Float16*)(ws + (size_t)179830784);
  _Float16* Wo16  = (_Float16*)(ws + (size_t)179879936);

  // 1) weight conversion to f16
  cvt_f16<<<(Etot * Cc + 255) / 256, 256, 0, stream>>>(W_in,  Win16, Etot * Cc);
  cvt_f16<<<(DBLW * Din + 255) / 256, 256, 0, stream>>>(W_x,  Wx16,  DBLW * Din);
  cvt_f16<<<(Cc * Din + 255) / 256, 256, 0, stream>>>(W_out, Wo16,  Cc * Din);

  // 2) LayerNorm -> xn (token-major f16)
  ln_kernel<<<Mtot, Cc, 0, stream>>>(x, ln_w, ln_b, xn16);

  // 3) in_proj GEMM: [16384,256] x [256,1024] -> xz f32
  wmma_gemm_store<4, Cc><<<dim3(Mtot / 16, Etot / 64), 32, 0, stream>>>(
      xn16, Win16, xz32, Etot);

  // 4) causal conv + SiLU -> xc (f32 + f16)
  conv_silu<<<(Mtot * Din) / 256, 256, 0, stream>>>(xz32, conv_w, conv_b, xc32, xc16);

  // 5) x_proj GEMM: [16384,512] x [512,48] -> dbl
  wmma_gemm_store<3, Din><<<dim3(Mtot / 16, 1), 32, 0, stream>>>(
      xc16, Wx16, dbl32, DBLW);

  // 6) delta = softplus(dt @ W_dt^T + b_dt)
  delta_kernel<<<(Mtot * Din) / 256, 256, 0, stream>>>(dbl32, W_dt, b_dt, dlt32);

  // 7) selective scan + D skip + silu(z) gate -> y f16
  scan_kernel<<<(4 * Din) / 16, 256, 0, stream>>>(
      dlt32, xc32, dbl32, xz32, A_log, Dp, y16);

  // 8) out_proj GEMM with transpose-store + residual
  wmma_outproj<<<dim3(Mtot / 16, Cc / 64), 32, 0, stream>>>(y16, Wo16, x, out);
}
